// ReBasedLinearAttention_23983097380980
// MI455X (gfx1250) — matrix operand order repaired and checked by reading, not compiled, not run on hardware
//
#include <hip/hip_runtime.h>
#include <hip/hip_bf16.h>
#include <cstdint>
#include <cstddef>

typedef __attribute__((ext_vector_type(2)))  float        v2f;
typedef __attribute__((ext_vector_type(8)))  float        v8f;
typedef __attribute__((ext_vector_type(2)))  _Float16     v2h;
typedef __attribute__((ext_vector_type(8)))  _Float16     v8h;
typedef __attribute__((ext_vector_type(16))) _Float16     v16h;
typedef __attribute__((ext_vector_type(4)))  unsigned int v4u;
typedef __attribute__((ext_vector_type(8)))  int          v8i;
typedef __attribute__((ext_vector_type(4)))  int          v4i;

#define EPSF 1e-5f

// ---------------------------------------------------------------------------
// Tensor Data Mover 2D tile copy (Global <-> LDS), D# per CDNA5 ISA.
// data_size = 4 bytes. Issue from one wave; tracked with TENSORcnt.
// ---------------------------------------------------------------------------
__device__ static inline void tdm_copy_2d(uint32_t lds_byte_addr, uint64_t gaddr,
                                          uint32_t width_elts, uint32_t height,
                                          uint32_t stride_elts, bool is_store) {
  v4u g0;
  g0[0] = 1u;                                           // count=1, user mode
  g0[1] = lds_byte_addr;                                // D#.lds_addr
  g0[2] = (uint32_t)gaddr;                              // global_addr[31:0]
  g0[3] = (uint32_t)((gaddr >> 32) & 0x01ffffffu)       // global_addr[56:32]
          | (2u << 30);                                 // type = 2 ("image")
  v8i g1;
  g1[0] = (int)(2u << 16);                              // data_size = 4B
  g1[1] = (int)(width_elts << 16);                      // tensor_dim0[15:0]
  g1[2] = (int)((width_elts >> 16) | (height << 16));   // dim0 hi | tensor_dim1 lo
  g1[3] = (int)((height >> 16) | (width_elts << 16));   // dim1 hi | tile_dim0
  g1[4] = (int)(height & 0xffffu);                      // tile_dim1 (tile_dim2=0)
  g1[5] = (int)stride_elts;                             // tensor_dim0_stride lo
  g1[6] = 0;
  g1[7] = 0;
  v4i z4 = {0, 0, 0, 0};
#if defined(__clang_major__) && __clang_major__ >= 23
  v8i z8 = {0, 0, 0, 0, 0, 0, 0, 0};
  if (is_store) __builtin_amdgcn_tensor_store_from_lds(g0, g1, z4, z4, z8, 0);
  else          __builtin_amdgcn_tensor_load_to_lds  (g0, g1, z4, z4, z8, 0);
#else
  if (is_store) __builtin_amdgcn_tensor_store_from_lds(g0, g1, z4, z4, 0);
  else          __builtin_amdgcn_tensor_load_to_lds  (g0, g1, z4, z4, 0);
#endif
}

// ---------------------------------------------------------------------------
// Generic C[M,N] = A[M,K] @ B[K,N], f32 in/out, f16 WMMA compute, f32 accum.
// Block: 256 threads (8 waves), tile 128x64, K-step 32.
// ---------------------------------------------------------------------------
__global__ __launch_bounds__(256) void gemm_f16_wmma(
    const float* __restrict__ A, const float* __restrict__ B,
    float* __restrict__ C, int M, int N, int K) {
  __shared__ _Float16 Ah[128][32];   // row-major A tile (f16)
  __shared__ _Float16 Bt[64][32];    // transposed B tile: Bt[n][k]

  const int t    = threadIdx.x;
  const int lane = t & 31;
  const int wid  = t >> 5;
  const int lrow = lane & 15;
  const int hi   = lane >> 4;          // 0: lanes 0-15, 1: lanes 16-31
  const int m0   = blockIdx.y * 128;
  const int n0   = blockIdx.x * 64;
  const int waveM = wid >> 1;          // 0..3
  const int waveN = wid & 1;           // 0..1

  v8f acc[2][2] = {};

  for (int kb = 0; kb < K; kb += 32) {
    // A tile: load float2, convert, packed b32 LDS store
#pragma unroll
    for (int i = 0; i < 8; ++i) {
      int e = t + i * 256;             // pair index 0..2047
      int r = e >> 4;
      int c = (e & 15) * 2;
      float2 f = *(const float2*)&A[(size_t)(m0 + r) * K + kb + c];
      v2h p; p[0] = (_Float16)f.x; p[1] = (_Float16)f.y;
      *(v2h*)&Ah[r][c] = p;
    }
    // B tile transposed: pack along k
#pragma unroll
    for (int i = 0; i < 4; ++i) {
      int e  = t + i * 256;            // 0..1023
      int c  = e & 63;
      int kk = (e >> 6) * 2;
      v2h p;
      p[0] = (_Float16)B[(size_t)(kb + kk) * N + n0 + c];
      p[1] = (_Float16)B[(size_t)(kb + kk + 1) * N + n0 + c];
      *(v2h*)&Bt[c][kk] = p;
    }
    __syncthreads();

    v16h a[2], b[2];
#pragma unroll
    for (int ti = 0; ti < 2; ++ti) {
      // A 16x32 f16 fragment: lane<16 -> K{0..7,16..23}, lane>=16 -> K{8..15,24..31}
      const _Float16* ap = &Ah[waveM * 32 + ti * 16 + lrow][0];
      v8h alo = *(const v8h*)(ap + hi * 8);
      v8h ahi = *(const v8h*)(ap + 16 + hi * 8);
      // B 32x16 f16 fragment: lane<16 -> K{0..7,16..23}, lane>=16 -> K{8..15,24..31} (col = lrow)
      const _Float16* bp = &Bt[waveN * 32 + ti * 16 + lrow][0];
      v8h blo = *(const v8h*)(bp + hi * 8);
      v8h bhi = *(const v8h*)(bp + 16 + hi * 8);
#pragma unroll
      for (int j = 0; j < 8; ++j) {
        a[ti][j] = alo[j]; a[ti][8 + j] = ahi[j];
        b[ti][j] = blo[j]; b[ti][8 + j] = bhi[j];
      }
    }
#pragma unroll
    for (int ti = 0; ti < 2; ++ti)
#pragma unroll
      for (int tj = 0; tj < 2; ++tj)
        acc[ti][tj] = __builtin_amdgcn_wmma_f32_16x16x32_f16(
            false, a[ti], false, b[tj], (short)0, acc[ti][tj], false, false);
    __syncthreads();
  }

#pragma unroll
  for (int ti = 0; ti < 2; ++ti)
#pragma unroll
    for (int tj = 0; tj < 2; ++tj)
#pragma unroll
      for (int r = 0; r < 8; ++r) {
        int row = m0 + waveM * 32 + ti * 16 + r + hi * 8;
        int col = n0 + waveN * 32 + tj * 16 + lrow;
        C[(size_t)row * N + col] = acc[ti][tj][r];
      }
}

// ---------------------------------------------------------------------------
// Chunked rebased linear attention. One block per (b,h). 256 threads.
// State S[256][64] + z[256] live in LDS across all 32 chunks (fp32, exact).
// q/k/v chunk tiles DMA'd in by the TDM; o chunk DMA'd out (overlapped: the
// store of chunk ch completes under the s_wait_tensorcnt of chunk ch+1).
// All GEMMs via v_wmma_f32_16x16x4_f32; phi fragments built on the fly.
// ---------------------------------------------------------------------------
#define ATTN_SMEM_FLOATS (256*64 + 64*16 + 64*16 + 64*64 + 64*64 + 256 + 64 + 256)

__global__ __launch_bounds__(256) void rebased_attn_kernel(
    const float* __restrict__ Q, const float* __restrict__ Kp,
    const float* __restrict__ V, float* __restrict__ O) {
  extern __shared__ float smem[];
  float* S     = smem;               // [256][64] running state
  float* qi    = S + 256 * 64;       // [64][16] (scaled by f^-0.5 after DMA)
  float* ki    = qi + 64 * 16;       // [64][16]
  float* vi    = ki + 64 * 16;       // [64][64]
  float* Am    = vi + 64 * 64;       // [64][64]  A matrix, then reused as o-chunk
  float* z     = Am + 64 * 64;       // [256] running normalizer
  float* zrow  = z + 256;            // [64] per-row denominator (+eps)
  float* zpart = zrow + 64;          // [256] partials

  const int t    = threadIdx.x;
  const int lane = t & 31;
  const int wid  = t >> 5;
  const int lrow = lane & 15;
  const int hi   = lane >> 4;

  const int bh = blockIdx.x;         // b*16 + h
  const int bi = bh >> 4, hh = bh & 15;
  const float* qbase = Q  + (size_t)bi * 2048 * 256  + hh * 16;
  const float* kbase = Kp + (size_t)bi * 2048 * 256  + hh * 16;
  const float* vbase = V  + (size_t)bi * 2048 * 1024 + hh * 64;
  const float* obase = O  + (size_t)bi * 2048 * 1024 + hh * 64;

  // LDS byte addresses for the TDM descriptors (flat addr[31:0] == LDS addr)
  const uint32_t lds_qi = (uint32_t)(uintptr_t)qi;
  const uint32_t lds_ki = (uint32_t)(uintptr_t)ki;
  const uint32_t lds_vi = (uint32_t)(uintptr_t)vi;
  const uint32_t lds_am = (uint32_t)(uintptr_t)Am;

  for (int i = t; i < 256 * 64; i += 256) S[i] = 0.f;
  z[t] = 0.f;
  __syncthreads();

  for (int ch = 0; ch < 32; ++ch) {
    // ---- 1. TDM: DMA q/k/v chunk tiles into LDS; wait also covers the
    //         previous chunk's o-store (same wave, in-order TENSORcnt) ----
    if (wid == 0) {
      tdm_copy_2d(lds_qi, (uint64_t)(uintptr_t)(qbase + (size_t)(ch * 64) * 256),
                  16, 64, 256, false);
      tdm_copy_2d(lds_ki, (uint64_t)(uintptr_t)(kbase + (size_t)(ch * 64) * 256),
                  16, 64, 256, false);
      tdm_copy_2d(lds_vi, (uint64_t)(uintptr_t)(vbase + (size_t)(ch * 64) * 1024),
                  64, 64, 1024, false);
      __builtin_amdgcn_s_wait_tensorcnt(0);
    }
    __syncthreads();
    // q scale f^-0.5 = 0.25 (TDM copies raw bits)
    for (int i = t; i < 1024; i += 256) qi[i] *= 0.25f;
    __syncthreads();

    // ---- 2. A = (qi @ ki^T)^2 * tril, via WMMA f32 16x16x4 ----
    for (int s = 0; s < 2; ++s) {
      int tile = wid + s * 8;                  // 16 tiles over 8 waves
      int mt = tile >> 2, nt = tile & 3;
      v8f c = {};
      int arow = mt * 16 + lrow;
      int bcol = nt * 16 + lrow;
#pragma unroll
      for (int kb = 0; kb < 16; kb += 4) {
        int k0 = kb + hi * 2;
        v2f a = *(const v2f*)&qi[arow * 16 + k0];
        v2f b = *(const v2f*)&ki[bcol * 16 + k0];   // B = ki^T
        c = __builtin_amdgcn_wmma_f32_16x16x4_f32(false, a, false, b,
                                                  (short)0, c, false, false);
      }
#pragma unroll
      for (int r = 0; r < 8; ++r) {
        int row = mt * 16 + r + hi * 8;
        int col = nt * 16 + lrow;
        float v = c[r];
        Am[row * 64 + col] = (row >= col) ? v * v : 0.f;
      }
    }
    __syncthreads();

    // ---- 3. denominators: inter_z = phi_q . z (old z), intra_z = rowsum(A) ----
    {
      int row = t & 63, part = t >> 6;
      const float* qr = &qi[row * 16];
      float s1 = 0.f;
      for (int kk = part * 64; kk < part * 64 + 64; ++kk)
        s1 += qr[kk >> 4] * qr[kk & 15] * z[kk];
      float s2 = 0.f;
      for (int cc = 0; cc < 16; ++cc)
        s2 += Am[row * 64 + part * 16 + cc];
      zpart[t] = s1 + s2;
    }
    __syncthreads();
    if (t < 64)
      zrow[t] = zpart[t] + zpart[64 + t] + zpart[128 + t] + zpart[192 + t] + EPSF;
    __syncthreads();

    // ---- 4. o = phi_q @ S + A @ vi (reads old S) ----
    v8f oacc[2];
    for (int s = 0; s < 2; ++s) {
      int tile = wid + s * 8;
      int mt = tile >> 2, nt = tile & 3;
      int arow = mt * 16 + lrow;
      int bcol = nt * 16 + lrow;
      const float* qr = &qi[arow * 16];
      v8f c = {};
#pragma unroll 4
      for (int kb = 0; kb < 256; kb += 4) {        // phi_q @ S, K = 256
        int k0 = kb + hi * 2;
        v2f a, b;
        a[0] = qr[k0 >> 4] * qr[k0 & 15];          // phi built on the fly
        a[1] = qr[(k0 + 1) >> 4] * qr[(k0 + 1) & 15];
        b[0] = S[k0 * 64 + bcol];
        b[1] = S[(k0 + 1) * 64 + bcol];
        c = __builtin_amdgcn_wmma_f32_16x16x4_f32(false, a, false, b,
                                                  (short)0, c, false, false);
      }
#pragma unroll 4
      for (int kb = 0; kb < 64; kb += 4) {         // A @ vi, K = 64
        int k0 = kb + hi * 2;
        v2f a = *(const v2f*)&Am[arow * 64 + k0];
        v2f b;
        b[0] = vi[k0 * 64 + bcol];
        b[1] = vi[(k0 + 1) * 64 + bcol];
        c = __builtin_amdgcn_wmma_f32_16x16x4_f32(false, a, false, b,
                                                  (short)0, c, false, false);
      }
      oacc[s] = c;
    }
    __syncthreads();   // all reads of old S / z / Am complete

    // ---- 5. S += phi_k^T @ vi (WMMA, accumulate in LDS-resident state) ----
    for (int s = 0; s < 8; ++s) {
      int tile = wid * 8 + s;                      // 64 tiles over 8 waves
      int mt = tile >> 2, nt = tile & 3;           // mt: f2 tile, nt: hd tile
      int col = nt * 16 + lrow;
      v8f c;
#pragma unroll
      for (int r = 0; r < 8; ++r) c[r] = S[(mt * 16 + r + hi * 8) * 64 + col];
      int m  = mt * 16 + lrow;                     // f2 row index
      int iA = m >> 4, jA = m & 15;
#pragma unroll 4
      for (int kb = 0; kb < 64; kb += 4) {
        int k0 = kb + hi * 2;
        v2f a, b;
        a[0] = ki[k0 * 16 + iA] * ki[k0 * 16 + jA];
        a[1] = ki[(k0 + 1) * 16 + iA] * ki[(k0 + 1) * 16 + jA];
        b[0] = vi[k0 * 64 + col];
        b[1] = vi[(k0 + 1) * 64 + col];
        c = __builtin_amdgcn_wmma_f32_16x16x4_f32(false, a, false, b,
                                                  (short)0, c, false, false);
      }
#pragma unroll
      for (int r = 0; r < 8; ++r) S[(mt * 16 + r + hi * 8) * 64 + col] = c[r];
    }
    // ---- z += column sums of phi_k ----
    {
      int iA = t >> 4, jA = t & 15;
      float s = 0.f;
      for (int c2 = 0; c2 < 64; ++c2) s += ki[c2 * 16 + iA] * ki[c2 * 16 + jA];
      z[t] += s;
    }

    // ---- 6. normalize into LDS o-chunk (reuse Am), then TDM store ----
    for (int s = 0; s < 2; ++s) {
      int tile = wid + s * 8;
      int mt = tile >> 2, nt = tile & 3;
#pragma unroll
      for (int r = 0; r < 8; ++r) {
        int row = mt * 16 + r + hi * 8;
        int col = nt * 16 + lrow;
        Am[row * 64 + col] = oacc[s][r] / zrow[row];
      }
    }
    __syncthreads();   // o-chunk + S/z updates visible
    if (wid == 0)      // async store; completion enforced by next chunk's wait
      tdm_copy_2d(lds_am, (uint64_t)(uintptr_t)(obase + (size_t)(ch * 64) * 1024),
                  64, 64, 1024, true);
  }
  if (wid == 0) __builtin_amdgcn_s_wait_tensorcnt(0);  // drain final o-store
}

// ---------------------------------------------------------------------------
extern "C" void kernel_launch(void* const* d_in, const int* in_sizes, int n_in,
                              void* d_out, int out_size, void* d_ws, size_t ws_size,
                              hipStream_t stream) {
  const float* hs = (const float*)d_in[0];   // [4,2048,1024]
  const float* Wq = (const float*)d_in[1];   // [1024,256]
  const float* Wk = (const float*)d_in[2];   // [1024,256]
  const float* Wv = (const float*)d_in[3];   // [1024,1024]
  const float* Wo = (const float*)d_in[4];   // [1024,1024]
  float* out = (float*)d_out;                // [4,2048,1024]

  const int M = 4 * 2048;                    // 8192 tokens
  float* ws  = (float*)d_ws;
  float* Qws = ws;                            // 8192*256
  float* Kws = Qws + (size_t)M * 256;         // 8192*256
  float* Vws = Kws + (size_t)M * 256;         // 8192*1024
  float* Ows = Vws + (size_t)M * 1024;        // 8192*1024

  // projections (f16 WMMA, f32 accumulate)
  gemm_f16_wmma<<<dim3(256 / 64,  M / 128), 256, 0, stream>>>(hs, Wq, Qws, M, 256, 1024);
  gemm_f16_wmma<<<dim3(256 / 64,  M / 128), 256, 0, stream>>>(hs, Wk, Kws, M, 256, 1024);
  gemm_f16_wmma<<<dim3(1024 / 64, M / 128), 256, 0, stream>>>(hs, Wv, Vws, M, 1024, 1024);

  // chunked rebased linear attention (exact fp32 state via f32 WMMA + TDM)
  const size_t attn_smem = (size_t)ATTN_SMEM_FLOATS * sizeof(float);  // ~106 KB
  hipFuncSetAttribute((const void*)rebased_attn_kernel,
                      hipFuncAttributeMaxDynamicSharedMemorySize, (int)attn_smem);
  rebased_attn_kernel<<<64, 256, attn_smem, stream>>>(Qws, Kws, Vws, Ows);

  // output projection
  gemm_f16_wmma<<<dim3(1024 / 64, M / 128), 256, 0, stream>>>(Ows, Wo, out, M, 1024, 1024);
}
